// seg_head_Deform_62311385531081
// MI455X (gfx1250) — matrix operand order repaired and checked by reading, not compiled, not run on hardware
//
#include <hip/hip_runtime.h>

// ---------------- problem constants ----------------
#define BB      2
#define CC      256
#define NHH     8
#define NLL     4
#define NPP     4
#define DHH     32
#define FFD     1024
#define NLAY    6
#define LQ      5440        // 4096+1024+256+64
#define MTOK    (BB*LQ)     // 10880 = 85*128

typedef __attribute__((ext_vector_type(16))) __bf16 v16bf;
typedef __attribute__((ext_vector_type(8)))  float  v8f;

struct U4 { unsigned int a, b, c, d; };          // POD 128-bit chunk
union FragBF { U4 q[2]; v16bf v; };              // 32B = 16 x bf16

// gfx1250 async global->LDS copy (ASYNCcnt path), guarded so both the host
// pass and toolchains without the builtin fall back to load+ds_store staging.
// Probe-learned signature: (int4 addrspace(1)*, int4 addrspace(3)*, Ii, Ii).
#if __has_builtin(__builtin_amdgcn_global_load_async_to_lds_b128) && \
    __has_builtin(__builtin_amdgcn_s_wait_asynccnt)
#define USE_ASYNC_LDS 1
typedef int vint4 __attribute__((vector_size(16)));
typedef __attribute__((address_space(1))) vint4 g_v4i;
typedef __attribute__((address_space(3))) vint4 l_v4i;
#else
#define USE_ASYNC_LDS 0
#endif

__device__ __forceinline__ unsigned short f2bf(float f) {
    unsigned int u = __float_as_uint(f);
    unsigned int r = u + 0x7fffu + ((u >> 16) & 1u);   // round-to-nearest-even
    return (unsigned short)(r >> 16);
}

// ---------------- fp32 -> bf16 convert ----------------
__global__ __launch_bounds__(256) void k_f2bf(const float* __restrict__ in,
                                              unsigned short* __restrict__ out, int n) {
    int i = blockIdx.x * 256 + threadIdx.x;
    if (i < n) out[i] = f2bf(in[i]);
}

// ---------------- feature transpose [B,CIN,HW] -> [B,HW,CIN] bf16 ----------------
__global__ __launch_bounds__(256) void k_featT(const float* __restrict__ f,
                                               unsigned short* __restrict__ o, int HW) {
    long long i = (long long)blockIdx.x * 256 + threadIdx.x;
    if (i >= (long long)BB * HW * CC) return;
    int b   = (int)(i / ((long long)HW * CC));
    int rem = (int)(i - (long long)b * HW * CC);
    int pix = rem / CC, ci = rem - pix * CC;
    o[i] = f2bf(f[((long long)b * CC + ci) * HW + pix]);
}

// ---------------- WMMA GEMM: Y[M,N] = X[M,K] * W[N,K]^T + bias ----------------
// block tile 128x128, 8 waves (4 down x 2 across), each wave 32x64
// (2x4 16x16 WMMA tiles), K-step 32. Staging: async global->LDS when available.
__global__ __launch_bounds__(256) void k_gemm_bf16(
    const unsigned short* __restrict__ X,    // [M,K] bf16
    const unsigned short* __restrict__ W,    // [N,K] bf16
    const float* __restrict__ bias,          // [N]
    float* __restrict__ Yf,                  // fp32 out (or null)
    unsigned short* __restrict__ Yb,         // bf16 out (or null)
    int M, int N, int K, int relu)
{
    __shared__ unsigned short lA[128][40];   // +8 pad, rows stay 16B aligned
    __shared__ unsigned short lB[128][40];

    const int tid  = threadIdx.x;
    const int lane = tid & 31, wid = tid >> 5;
    const int m0 = blockIdx.y * 128;
    const int n0 = blockIdx.x * 128;
    const int wm = (wid >> 1) * 32;          // wave M offset (4 waves down)
    const int wn = (wid & 1) * 64;           // wave N offset (2 waves across)

    v8f acc[2][4];
    #pragma unroll
    for (int i = 0; i < 2; ++i)
        #pragma unroll
        for (int j = 0; j < 4; ++j)
            acc[i][j] = (v8f){0.f,0.f,0.f,0.f,0.f,0.f,0.f,0.f};

    const int aRow = wm + (lane & 15);
    const int koffA = (lane >> 4) * 8;       // A: interleaved 8-elem K runs (ISA layout)
    const int bCol = wn + (lane & 15);
    const int kbB  = (lane >> 4) * 8;        // B: interleaved 8-elem K runs (ISA layout)

    for (int k0 = 0; k0 < K; k0 += 32) {
        __syncthreads();                     // protect LDS WAR vs previous ds_loads
        #pragma unroll
        for (int it = 0; it < 2; ++it) {
            int li = tid + it * 256;
            int r = li >> 2, ch = li & 3;    // A: 128 rows x 4 chunks of 8 bf16
            const unsigned short* ga = X + (long long)(m0 + r) * K + k0 + ch * 8;
            const unsigned short* gb = W + (long long)(n0 + r) * K + k0 + ch * 8;
#if USE_ASYNC_LDS
            __builtin_amdgcn_global_load_async_to_lds_b128(
                (g_v4i*)ga, (l_v4i*)&lA[r][ch * 8], 0, 0);
            __builtin_amdgcn_global_load_async_to_lds_b128(
                (g_v4i*)gb, (l_v4i*)&lB[r][ch * 8], 0, 0);
#else
            *(U4*)&lA[r][ch * 8] = *(const U4*)ga;
            *(U4*)&lB[r][ch * 8] = *(const U4*)gb;
#endif
        }
        if (k0 + 32 < K) {   // pull next K tile toward the WGP (global_prefetch_b8)
            __builtin_prefetch(X + (long long)(m0 + (tid >> 1)) * K + k0 + 32, 0, 1);
            __builtin_prefetch(W + (long long)(n0 + (tid >> 1)) * K + k0 + 32, 0, 1);
        }
#if USE_ASYNC_LDS
        __builtin_amdgcn_s_wait_asynccnt(0); // our async writes landed in LDS
#endif
        __syncthreads();                     // all waves' tiles visible

        #pragma unroll
        for (int i = 0; i < 2; ++i) {
            FragBF af;
            af.q[0] = *(const U4*)&lA[aRow + i * 16][koffA];
            af.q[1] = *(const U4*)&lA[aRow + i * 16][koffA + 16];
            #pragma unroll
            for (int j = 0; j < 4; ++j) {
                FragBF bf;
                bf.q[0] = *(const U4*)&lB[bCol + j * 16][kbB];
                bf.q[1] = *(const U4*)&lB[bCol + j * 16][kbB + 16];
                acc[i][j] = __builtin_amdgcn_wmma_f32_16x16x32_bf16(
                    false, af.v, false, bf.v, (short)0, acc[i][j], false, false);
            }
        }
    }

    // D layout: lane -> N = lane&15 ; VGPR r -> M = r + 8*(lane>>4)
    #pragma unroll
    for (int i = 0; i < 2; ++i)
        #pragma unroll
        for (int j = 0; j < 4; ++j) {
            int col  = n0 + wn + j * 16 + (lane & 15);
            int rowb = m0 + wm + i * 16 + (lane >> 4) * 8;
            float bs = bias ? bias[col] : 0.f;
            #pragma unroll
            for (int r = 0; r < 8; ++r) {
                float v = acc[i][j][r] + bs;
                if (relu) v = fmaxf(v, 0.f);
                long long o = (long long)(rowb + r) * N + col;
                if (Yf) Yf[o] = v;
                if (Yb) Yb[o] = f2bf(v);
            }
        }
}

// ---------------- GroupNorm (32 groups) -> token-major fp32 + bf16 ----------------
__global__ __launch_bounds__(256) void k_groupnorm(
    const float* __restrict__ pin,           // [B,HW,256]
    const float* __restrict__ gw, const float* __restrict__ gb, // [256] (level slice)
    float* __restrict__ xf, unsigned short* __restrict__ xb,
    int HW, int start)
{
    __shared__ float s1[256], s2[256];
    int b = blockIdx.x >> 5, g = blockIdx.x & 31;
    int tid = threadIdx.x;
    int n = HW * 8;
    float a1 = 0.f, a2 = 0.f;
    for (int i = tid; i < n; i += 256) {
        int pix = i >> 3, cc = i & 7;
        float v = pin[((long long)b * HW + pix) * CC + g * 8 + cc];
        a1 += v; a2 += v * v;
    }
    s1[tid] = a1; s2[tid] = a2; __syncthreads();
    for (int st = 128; st > 0; st >>= 1) {
        if (tid < st) { s1[tid] += s1[tid + st]; s2[tid] += s2[tid + st]; }
        __syncthreads();
    }
    float mu = s1[0] / n, var = s2[0] / n - mu * mu;
    float rs = rsqrtf(var + 1e-5f);
    for (int i = tid; i < n; i += 256) {
        int pix = i >> 3, cc = i & 7, ch = g * 8 + cc;
        float v = (pin[((long long)b * HW + pix) * CC + ch] - mu) * rs * gw[ch] + gb[ch];
        long long o = ((long long)b * LQ + start + pix) * CC + ch;
        xf[o] = v; xb[o] = f2bf(v);
    }
}

// ---------------- sine positional embedding + level embed ----------------
__global__ __launch_bounds__(256) void k_pos(const float* __restrict__ lemb,
                                             float* __restrict__ pos,
                                             int h, int w, int start)
{
    int pix = blockIdx.x, c = threadIdx.x;
    int iy = pix / w, ix = pix - iy * w;
    const float TWO_PI = 6.2831853071795864f;
    int cc = (c < 128) ? c : c - 128;
    float v = (c < 128) ? ((iy + 1) / ((float)h + 1e-6f)) * TWO_PI
                        : ((ix + 1) / ((float)w + 1e-6f)) * TWO_PI;
    float t = powf(10000.f, (float)(2 * (cc >> 1)) / 128.f);
    float p = v / t;
    float val = (cc & 1) ? cosf(p) : sinf(p);
    pos[(long long)(start + pix) * CC + c] = val + lemb[c];
}

__global__ void k_ref(float* __restrict__ ref, int h, int w, int start) {
    int pix = blockIdx.x * 256 + threadIdx.x;
    if (pix >= h * w) return;
    int iy = pix / w, ix = pix - iy * w;
    ref[(start + pix) * 2 + 0] = (ix + 0.5f) / w;
    ref[(start + pix) * 2 + 1] = (iy + 0.5f) / h;
}

// ---------------- q = bf16(x + pos) ----------------
__global__ __launch_bounds__(256) void k_addpos(const float* __restrict__ x,
                                                const float* __restrict__ pos,
                                                unsigned short* __restrict__ q) {
    long long i = (long long)blockIdx.x * 256 + threadIdx.x;
    long long tc = i % ((long long)LQ * CC);
    q[i] = f2bf(x[i] + pos[tc]);
}

// ---------------- MS-deform sampling: one wave per (b,q,head); lane = channel ----
__global__ __launch_bounds__(256) void k_msda(
    const float* __restrict__ off, const float* __restrict__ aw,
    const float* __restrict__ value, const float* __restrict__ ref,
    unsigned short* __restrict__ samp)
{
    const int wid = threadIdx.x >> 5, lane = threadIdx.x & 31;
    int gw = blockIdx.x * 8 + wid;
    if (gw >= BB * LQ * NHH) return;
    int b   = gw / (LQ * NHH);
    int rem = gw - b * (LQ * NHH);
    int qi  = rem / NHH;
    int h   = rem - qi * NHH;

    const int HWs[4] = {64, 32, 16, 8};
    const int ST[4]  = {0, 4096, 5120, 5376};

    // softmax over NL*NP = 16 (joint across levels, as in reference)
    const float* awp = aw + (((long long)b * LQ + qi) * NHH + h) * 16;
    float lg[16], mx = -1e30f;
    #pragma unroll
    for (int t = 0; t < 16; ++t) { lg[t] = awp[t]; mx = fmaxf(mx, lg[t]); }
    float s = 0.f;
    #pragma unroll
    for (int t = 0; t < 16; ++t) { lg[t] = expf(lg[t] - mx); s += lg[t]; }
    float inv = 1.f / s;

    const float rx = ref[qi * 2 + 0], ry = ref[qi * 2 + 1];
    const float* offp = off + (((long long)b * LQ + qi) * NHH + h) * (NLL * NPP * 2);
    float accv = 0.f;

    #pragma unroll
    for (int l = 0; l < 4; ++l) {
        const int Wl = HWs[l], Hl = HWs[l];
        const float* vl = value + ((long long)b * LQ + ST[l]) * CC + h * DHH + lane;
        #pragma unroll
        for (int p = 0; p < 4; ++p) {
            float ox = offp[(l * 4 + p) * 2 + 0];
            float oy = offp[(l * 4 + p) * 2 + 1];
            // loc = ref + off/[W,H]; pixel coords = loc*W - 0.5
            float x = rx * Wl + ox - 0.5f;
            float y = ry * Hl + oy - 0.5f;
            float x0f = floorf(x), y0f = floorf(y);
            float fx = x - x0f, fy = y - y0f;
            int x0 = (int)x0f, y0 = (int)y0f;
            float wgt = lg[l * 4 + p] * inv;
            float cw[4] = {(1.f - fx) * (1.f - fy), fx * (1.f - fy),
                           (1.f - fx) * fy,         fx * fy};
            int dx[4] = {0, 1, 0, 1}, dy[4] = {0, 0, 1, 1};
            #pragma unroll
            for (int cidx = 0; cidx < 4; ++cidx) {
                int xi = x0 + dx[cidx], yi = y0 + dy[cidx];
                if (xi >= 0 && xi < Wl && yi >= 0 && yi < Hl)
                    accv += wgt * cw[cidx] * vl[(long long)(yi * Wl + xi) * CC];
            }
        }
    }
    samp[((long long)b * LQ + qi) * CC + h * DHH + lane] = f2bf(accv);
}

// ---------------- residual add + LayerNorm (writes fp32 + bf16) ----------------
__global__ __launch_bounds__(256) void k_add_ln(
    const float* __restrict__ xin, const float* __restrict__ delta,
    const float* __restrict__ w, const float* __restrict__ bb,
    float* __restrict__ xout, unsigned short* __restrict__ xbf)
{
    __shared__ float s1[256], s2[256];
    long long t = blockIdx.x;
    int c = threadIdx.x;
    float v = xin[t * CC + c] + delta[t * CC + c];
    s1[c] = v; s2[c] = v * v; __syncthreads();
    for (int st = 128; st > 0; st >>= 1) {
        if (c < st) { s1[c] += s1[c + st]; s2[c] += s2[c + st]; }
        __syncthreads();
    }
    float mu = s1[0] * (1.f / CC);
    float var = s2[0] * (1.f / CC) - mu * mu;
    float y = (v - mu) * rsqrtf(var + 1e-5f) * w[c] + bb[c];
    xout[t * CC + c] = y;
    xbf[t * CC + c] = f2bf(y);
}

// ---------------- head convs + bilinear resize ----------------
__global__ __launch_bounds__(256) void k_conv1(
    const float* __restrict__ x,  // tokens [B,LQ,256], level0 at offset 0 => [64,64]
    const float* __restrict__ wgt, const float* __restrict__ bias,
    float* __restrict__ y)        // [B,64,64,64]
{
    int idx = blockIdx.x * 256 + threadIdx.x;      // B*64*64*64
    int ox = idx & 63, oy = (idx >> 6) & 63, oc = (idx >> 12) & 63, b = idx >> 18;
    float acc = bias[oc];
    for (int ky = 0; ky < 3; ++ky) {
        int iy = oy + ky - 1; if ((unsigned)iy >= 64u) continue;
        for (int kx = 0; kx < 3; ++kx) {
            int ix = ox + kx - 1; if ((unsigned)ix >= 64u) continue;
            const float* xp = x + ((long long)b * LQ + iy * 64 + ix) * CC;
            const float* wp = wgt + (long long)oc * CC * 9 + ky * 3 + kx;
            for (int ci = 0; ci < CC; ++ci) acc += xp[ci] * wp[(long long)ci * 9];
        }
    }
    y[idx] = fmaxf(acc, 0.f);
}

__global__ __launch_bounds__(256) void k_conv2(
    const float* __restrict__ y1, const float* __restrict__ wgt,
    const float* __restrict__ bias, float* __restrict__ y2)   // [B,2,64,64]
{
    int idx = blockIdx.x * 256 + threadIdx.x;      // B*2*4096
    int ox = idx & 63, oy = (idx >> 6) & 63, oc = (idx >> 12) & 1, b = idx >> 13;
    float acc = bias[oc];
    for (int ky = 0; ky < 3; ++ky) {
        int iy = oy + ky - 1; if ((unsigned)iy >= 64u) continue;
        for (int kx = 0; kx < 3; ++kx) {
            int ix = ox + kx - 1; if ((unsigned)ix >= 64u) continue;
            for (int ci = 0; ci < 64; ++ci)
                acc += y1[(((long long)b * 64 + ci) * 64 + iy) * 64 + ix]
                     * wgt[((long long)oc * 64 + ci) * 9 + ky * 3 + kx];
        }
    }
    y2[idx] = acc;
}

__global__ __launch_bounds__(256) void k_resize(const float* __restrict__ y2,
                                                float* __restrict__ out) {
    int idx = blockIdx.x * 256 + threadIdx.x;      // B*2*256*256
    int ox = idx & 255, oy = (idx >> 8) & 255, oc = (idx >> 16) & 1, b = idx >> 17;
    float sx = (ox + 0.5f) * 0.25f - 0.5f;
    float sy = (oy + 0.5f) * 0.25f - 0.5f;
    float x0f = floorf(sx), y0f = floorf(sy);
    float fx = sx - x0f, fy = sy - y0f;
    int x0 = max((int)x0f, 0), y0 = max((int)y0f, 0);
    int x1 = min((int)x0f + 1, 63), y1 = min((int)y0f + 1, 63);
    const float* p = y2 + ((long long)(b * 2 + oc)) * 4096;
    float v = (1.f - fx) * (1.f - fy) * p[y0 * 64 + x0]
            + fx * (1.f - fy) * p[y0 * 64 + x1]
            + (1.f - fx) * fy * p[y1 * 64 + x0]
            + fx * fy * p[y1 * 64 + x1];
    out[idx] = v;
}

// ============================== host launcher ==============================
static inline size_t alignup(size_t x) { return (x + 255) & ~(size_t)255; }

extern "C" void kernel_launch(void* const* d_in, const int* in_sizes, int n_in,
                              void* d_out, int out_size, void* d_ws, size_t ws_size,
                              hipStream_t stream) {
    (void)in_sizes; (void)n_in; (void)out_size; (void)ws_size;

    // ---- inputs (setup_inputs order) ----
    const float* feats[4] = {(const float*)d_in[0], (const float*)d_in[1],
                             (const float*)d_in[2], (const float*)d_in[3]};
    const float* proj_w = (const float*)d_in[4];
    const float* proj_b = (const float*)d_in[5];
    const float* gn_w   = (const float*)d_in[6];
    const float* gn_b   = (const float*)d_in[7];
    const float* lemb   = (const float*)d_in[8];
    const float* so_w   = (const float*)d_in[9];
    const float* so_b   = (const float*)d_in[10];
    const float* aw_w   = (const float*)d_in[11];
    const float* aw_b   = (const float*)d_in[12];
    const float* vp_w   = (const float*)d_in[13];
    const float* vp_b   = (const float*)d_in[14];
    const float* op_w   = (const float*)d_in[15];
    const float* op_b   = (const float*)d_in[16];
    const float* n1_w   = (const float*)d_in[17];
    const float* n1_b   = (const float*)d_in[18];
    const float* l1_w   = (const float*)d_in[19];
    const float* l1_b   = (const float*)d_in[20];
    const float* l2_w   = (const float*)d_in[21];
    const float* l2_b   = (const float*)d_in[22];
    const float* n2_w   = (const float*)d_in[23];
    const float* n2_b   = (const float*)d_in[24];
    const float* c1_w   = (const float*)d_in[25];
    const float* c1_b   = (const float*)d_in[26];
    const float* c2_w   = (const float*)d_in[27];
    const float* c2_b   = (const float*)d_in[28];

    // ---- workspace bump allocator ----
    char* base = (char*)d_ws;
    size_t off = 0;
    auto A = [&](size_t bytes) -> void* { void* p = base + off; off = alignup(off + bytes); return p; };

    unsigned short* wb_proj = (unsigned short*)A((size_t)NLL * CC * CC * 2);
    unsigned short* wb_so   = (unsigned short*)A((size_t)NLAY * 256 * CC * 2);
    unsigned short* wb_aw   = (unsigned short*)A((size_t)NLAY * 128 * CC * 2);
    unsigned short* wb_vp   = (unsigned short*)A((size_t)NLAY * CC * CC * 2);
    unsigned short* wb_op   = (unsigned short*)A((size_t)NLAY * CC * CC * 2);
    unsigned short* wb_l1   = (unsigned short*)A((size_t)NLAY * FFD * CC * 2);
    unsigned short* wb_l2   = (unsigned short*)A((size_t)NLAY * CC * FFD * 2);
    float* pos  = (float*)A((size_t)LQ * CC * 4);
    float* refp = (float*)A((size_t)LQ * 2 * 4);
    float* xf   = (float*)A((size_t)MTOK * CC * 4);
    unsigned short* xb    = (unsigned short*)A((size_t)MTOK * CC * 2);
    unsigned short* qb    = (unsigned short*)A((size_t)MTOK * CC * 2);
    float* val  = (float*)A((size_t)MTOK * CC * 4);
    float* offb = (float*)A((size_t)MTOK * 256 * 4);
    float* awb  = (float*)A((size_t)MTOK * 128 * 4);
    unsigned short* sampb = (unsigned short*)A((size_t)MTOK * CC * 2);
    float* tmp  = (float*)A((size_t)MTOK * CC * 4);
    unsigned short* ffb   = (unsigned short*)A((size_t)MTOK * FFD * 2);
    unsigned short* ftT   = (unsigned short*)A((size_t)BB * 4096 * CC * 2);
    float* pout = (float*)A((size_t)BB * 4096 * CC * 4);
    float* y1   = (float*)A((size_t)BB * 64 * 4096 * 4);
    float* y2   = (float*)A((size_t)BB * 2 * 4096 * 4);

    auto cvt = [&](const float* src, unsigned short* dst, int n) {
        k_f2bf<<<(n + 255) / 256, 256, 0, stream>>>(src, dst, n);
    };
    cvt(proj_w, wb_proj, NLL * CC * CC);
    cvt(so_w,   wb_so,   NLAY * 256 * CC);
    cvt(aw_w,   wb_aw,   NLAY * 128 * CC);
    cvt(vp_w,   wb_vp,   NLAY * CC * CC);
    cvt(op_w,   wb_op,   NLAY * CC * CC);
    cvt(l1_w,   wb_l1,   NLAY * FFD * CC);
    cvt(l2_w,   wb_l2,   NLAY * CC * FFD);

    auto gemm = [&](const unsigned short* X, const unsigned short* W,
                    const float* bias, float* Yf, unsigned short* Yb,
                    int M, int N, int K, int relu) {
        dim3 grid(N / 128, M / 128);
        k_gemm_bf16<<<grid, 256, 0, stream>>>(X, W, bias, Yf, Yb, M, N, K, relu);
    };

    // ---- input proj + GroupNorm + pos/ref per level ----
    const int Hs[4] = {64, 32, 16, 8};
    const int ST[4] = {0, 4096, 5120, 5376};
    for (int l = 0; l < 4; ++l) {
        int HW = Hs[l] * Hs[l];
        k_featT<<<(BB * HW * CC + 255) / 256, 256, 0, stream>>>(feats[l], ftT, HW);
        gemm(ftT, wb_proj + (size_t)l * CC * CC, proj_b + l * CC,
             pout, nullptr, BB * HW, CC, CC, 0);
        k_groupnorm<<<BB * 32, 256, 0, stream>>>(pout, gn_w + l * CC, gn_b + l * CC,
                                                 xf, xb, HW, ST[l]);
        k_pos<<<HW, 256, 0, stream>>>(lemb + l * CC, pos, Hs[l], Hs[l], ST[l]);
        k_ref<<<(HW + 255) / 256, 256, 0, stream>>>(refp, Hs[l], Hs[l], ST[l]);
    }

    // ---- 6 encoder layers ----
    for (int i = 0; i < NLAY; ++i) {
        k_addpos<<<MTOK, 256, 0, stream>>>(xf, pos, qb);
        gemm(xb, wb_vp + (size_t)i * CC * CC, vp_b + i * CC, val, nullptr, MTOK, CC, CC, 0);
        gemm(qb, wb_so + (size_t)i * 256 * CC, so_b + i * 256, offb, nullptr, MTOK, 256, CC, 0);
        gemm(qb, wb_aw + (size_t)i * 128 * CC, aw_b + i * 128, awb, nullptr, MTOK, 128, CC, 0);
        k_msda<<<(BB * LQ * NHH) / 8, 256, 0, stream>>>(offb, awb, val, refp, sampb);
        gemm(sampb, wb_op + (size_t)i * CC * CC, op_b + i * CC, tmp, nullptr, MTOK, CC, CC, 0);
        k_add_ln<<<MTOK, 256, 0, stream>>>(xf, tmp, n1_w + i * CC, n1_b + i * CC, xf, xb);
        gemm(xb, wb_l1 + (size_t)i * FFD * CC, l1_b + i * FFD, nullptr, ffb, MTOK, FFD, CC, 1);
        gemm(ffb, wb_l2 + (size_t)i * CC * FFD, l2_b + i * CC, tmp, nullptr, MTOK, CC, FFD, 0);
        k_add_ln<<<MTOK, 256, 0, stream>>>(xf, tmp, n2_w + i * CC, n2_b + i * CC, xf, xb);
    }

    // ---- head: conv3x3(C->64)+relu, conv3x3(64->2), bilinear resize 64->256 ----
    k_conv1<<<(BB * 64 * 4096) / 256, 256, 0, stream>>>(xf, c1_w, c1_b, y1);
    k_conv2<<<(BB * 2 * 4096) / 256, 256, 0, stream>>>(y1, c2_w, c2_b, y2);
    k_resize<<<(BB * 2 * 256 * 256) / 256, 256, 0, stream>>>(y2, (float*)d_out);
}
